// Multi_head_Attn_85048942395647
// MI455X (gfx1250) — matrix operand order repaired and checked by reading, not compiled, not run on hardware
//
#include <hip/hip_runtime.h>
#include <hip/hip_bf16.h>
#include <math.h>

typedef __attribute__((ext_vector_type(16))) _Float16 v16h;
typedef __attribute__((ext_vector_type(8)))  float    v8f;

#define B_  16
#define L_  4096
#define DM_ 1024
#define NH_ 16
#define DV_ 64

// ---------------------------------------------------------------------------
// k0: convert w_v (1024x1024 fp32, row-major [k][n]) into hi/lo f16 fragments
// in a contiguous-k 32x16 B-fragment lane order (k5 stages A the same way):
//   frag index = ((s*64 + j)*32 + lane)*16 + t
//   K = 32*s + (lane&16) + t,  N = 16*j + (lane&15)
// ---------------------------------------------------------------------------
__global__ __launch_bounds__(256) void k0_convert_wv(
    const float* __restrict__ w_v, _Float16* __restrict__ bhi,
    _Float16* __restrict__ blo) {
  int idx  = blockIdx.x * 256 + threadIdx.x;   // 0 .. 1048575
  int t    = idx & 15;
  int lane = (idx >> 4) & 31;
  int j    = (idx >> 9) & 63;
  int s    = idx >> 15;
  int k = 32 * s + (lane & 16) + t;
  int n = 16 * j + (lane & 15);
  float v = w_v[k * DM_ + n];
  _Float16 hi = (_Float16)v;
  _Float16 lo = (_Float16)(v - (float)hi);
  bhi[idx] = hi;
  blo[idx] = lo;
}

// ---------------------------------------------------------------------------
// k1: q[b, j] = enc[b, L-1, :] . w_q[:, j] + b_q[j]
// grid: (32 jj, 16 b), block 256: j = jj*32 + lane, cpart = wave (8 K-slices)
// ---------------------------------------------------------------------------
__global__ __launch_bounds__(256) void k1_q(
    const float* __restrict__ enc, const float* __restrict__ w_q,
    const float* __restrict__ b_q, float* __restrict__ q) {
  int b  = blockIdx.y;
  int jj = blockIdx.x;
  int tid = threadIdx.x;
  int j     = jj * 32 + (tid & 31);
  int cpart = tid >> 5;
  const float* er = enc + ((long)b * L_ + (L_ - 1)) * DM_;
  float acc = 0.f;
  int c0 = cpart * 128;
  for (int c = c0; c < c0 + 128; ++c) acc += er[c] * w_q[(long)c * DM_ + j];
  __shared__ float red[8][32];
  red[cpart][tid & 31] = acc;
  __syncthreads();
  if (cpart == 0) {
    float s = b_q[j];
    #pragma unroll
    for (int p = 0; p < 8; ++p) s += red[p][tid & 31];
    q[b * DM_ + j] = s;
  }
}

// ---------------------------------------------------------------------------
// k2: wq_eff[b,h,c] = sum_d q[b,h*64+d] * w_k[c, h*64+d];   cterm[b,h] = q.b_k
// grid: (4 cc, 16 h, 16 b), block 256: c = cc*256 + tid
// ---------------------------------------------------------------------------
__global__ __launch_bounds__(256) void k2_wqeff(
    const float* __restrict__ w_k, const float* __restrict__ b_k,
    const float* __restrict__ q, float* __restrict__ wq_eff,
    float* __restrict__ cterm) {
  int cc = blockIdx.x, h = blockIdx.y, b = blockIdx.z;
  int tid = threadIdx.x;
  __shared__ float qs[DV_];
  if (tid < DV_) qs[tid] = q[b * DM_ + h * DV_ + tid];
  __syncthreads();
  int c = cc * 256 + tid;
  const float* wkp = w_k + (long)c * DM_ + h * DV_;
  float acc = 0.f;
  #pragma unroll 8
  for (int d = 0; d < DV_; ++d) acc += qs[d] * wkp[d];
  wq_eff[((b * NH_ + h) * DM_) + c] = acc;
  if (cc == 0 && tid == 0) {
    float ct = 0.f;
    for (int d = 0; d < DV_; ++d) ct += qs[d] * b_k[h * DV_ + d];
    cterm[b * NH_ + h] = ct;
  }
}

// ---------------------------------------------------------------------------
// k3: scores[(b*16+h)*L + l] = (enc[b,l,:] . wq_eff[b,h,:] + cterm) / (32+1e-5)
// grid: (512 rowblk, 16 b), block 256 = 8 waves, one row per wave.
// wq_eff[b] (64KB) staged in LDS.
// ---------------------------------------------------------------------------
__global__ __launch_bounds__(256) void k3_scores(
    const float* __restrict__ enc, const float* __restrict__ wq_eff,
    const float* __restrict__ cterm, float* __restrict__ scores) {
  __shared__ float wqs[NH_ * DM_];
  int b = blockIdx.y;
  int row = blockIdx.x * 8 + (threadIdx.x >> 5);
  int tid = threadIdx.x, lane = tid & 31;
  for (int i = tid; i < NH_ * DM_; i += 256) wqs[i] = wq_eff[b * NH_ * DM_ + i];
  __syncthreads();
  const float* er = enc + ((long)b * L_ + row) * DM_;
  float acc[NH_];
  #pragma unroll
  for (int h = 0; h < NH_; ++h) acc[h] = 0.f;
  for (int c = lane; c < DM_; c += 32) {
    float e = er[c];
    #pragma unroll
    for (int h = 0; h < NH_; ++h) acc[h] += e * wqs[h * DM_ + c];
  }
  const float inv_temp = 1.0f / (32.0f + 1e-5f);
  #pragma unroll
  for (int h = 0; h < NH_; ++h) {
    float v = acc[h];
    #pragma unroll
    for (int off = 16; off > 0; off >>= 1) v += __shfl_xor(v, off, 32);
    if (lane == 0)
      scores[((b * NH_ + h) * L_) + row] = (v + cterm[b * NH_ + h]) * inv_temp;
  }
}

// ---------------------------------------------------------------------------
// k4: in-place log_softmax over L per (b,h) row. grid 256, block 256.
// ---------------------------------------------------------------------------
__global__ __launch_bounds__(256) void k4_logsoftmax(float* __restrict__ sc) {
  float* row = sc + (long)blockIdx.x * L_;
  int tid = threadIdx.x, lane = tid & 31, wv = tid >> 5;
  float v[16];
  float mx = -1e30f;
  #pragma unroll
  for (int i = 0; i < 16; ++i) {
    v[i] = row[tid + 256 * i];
    mx = fmaxf(mx, v[i]);
  }
  #pragma unroll
  for (int off = 16; off > 0; off >>= 1) mx = fmaxf(mx, __shfl_xor(mx, off, 32));
  __shared__ float r1[8], r2[8];
  if (lane == 0) r1[wv] = mx;
  __syncthreads();
  mx = r1[0];
  #pragma unroll
  for (int i = 1; i < 8; ++i) mx = fmaxf(mx, r1[i]);
  float s = 0.f;
  #pragma unroll
  for (int i = 0; i < 16; ++i) s += expf(v[i] - mx);
  #pragma unroll
  for (int off = 16; off > 0; off >>= 1) s += __shfl_xor(s, off, 32);
  if (lane == 0) r2[wv] = s;
  __syncthreads();
  s = 0.f;
  #pragma unroll
  for (int i = 0; i < 8; ++i) s += r2[i];
  float lg = mx + logf(s);
  #pragma unroll
  for (int i = 0; i < 16; ++i) row[tid + 256 * i] = v[i] - lg;
}

// ---------------------------------------------------------------------------
// k5: fused  V = enc @ w_v (+b_v),  out = LN(attn*V + enc)
// grid 4096 (16-row tiles), block 512 = 16 waves. Wave w owns N in [64w,64w+64)
// (== head w). Split-f16 GEMM: 3 WMMAs per 16x16x32 step.
// The whole 16x1024 A tile is converted to hi/lo f16 ONCE and stored in LDS in
// the same contiguous-k fragment order as B, so the k-loop has no barriers and no conversion
// VALU: per step, each wave does 2x 32B LDS loads per A operand + 12 WMMAs.
// The 64KB fragment buffer is aliased with the small LN-partials region.
// ---------------------------------------------------------------------------
__global__ __launch_bounds__(512) void k5_vgemm_combine_ln(
    const float* __restrict__ enc, const _Float16* __restrict__ bhi,
    const _Float16* __restrict__ blo, const float* __restrict__ b_v,
    const float* __restrict__ attn, const float* __restrict__ ln_g,
    const float* __restrict__ ln_b, float* __restrict__ out) {
  __shared__ __align__(32) unsigned char smem[65536];
  _Float16* hifrag = (_Float16*)smem;            // 32 KB: [s][lane][16]
  _Float16* lofrag = (_Float16*)(smem + 32768);  // 32 KB: [s][lane][16]

  const int tid  = threadIdx.x;
  const int lane = tid & 31;
  const int w    = tid >> 5;                    // wave == head
  const long r0  = (long)blockIdx.x * 16;       // first global row
  const int  b   = (int)(r0 >> 12);
  const int  l0  = (int)(r0 & (L_ - 1));
  const int  mrow = lane >> 4;                  // C: 0 -> rows 0..7, 1 -> 8..15

  // ---- stage + convert the whole 16x1024 A tile into fragment order -------
  {
    const int r  = tid >> 5;        // row 0..15
    const int cl = tid & 31;        // col phase
    // fragment coords are constant per thread (col group repeats mod 32)
    const int half  = cl >> 4;
    const int lane2 = r + 16 * half;
    const int elem  = cl & 15;
    const float* erow = enc + (r0 + r) * DM_;
    #pragma unroll 4
    for (int i = 0; i < 32; ++i) {
      float x = erow[cl + 32 * i];
      _Float16 h = (_Float16)x;
      _Float16 l = (_Float16)(x - (float)h);
      int idx = (i * 32 + lane2) * 16 + elem;
      hifrag[idx] = h;
      lofrag[idx] = l;
    }
  }
  __syncthreads();

  // ---- main K loop ---------------------------------------------------------
  v8f acc[4] = {v8f{}, v8f{}, v8f{}, v8f{}};
  for (int s = 0; s < 32; ++s) {
    const v16h ahi = *(const v16h*)(hifrag + (s * 32 + lane) * 16);
    const v16h alo = *(const v16h*)(lofrag + (s * 32 + lane) * 16);
    const long bbase = (((long)s * 64 + w * 4) * 32 + lane) * 16;
    #pragma unroll
    for (int j = 0; j < 4; ++j) {
      v16h bh = *(const v16h*)(bhi + bbase + j * 512);
      v16h bl = *(const v16h*)(blo + bbase + j * 512);
      acc[j] = __builtin_amdgcn_wmma_f32_16x16x32_f16(false, ahi, false, bh,
                                                      (short)0, acc[j], false, false);
      acc[j] = __builtin_amdgcn_wmma_f32_16x16x32_f16(false, ahi, false, bl,
                                                      (short)0, acc[j], false, false);
      acc[j] = __builtin_amdgcn_wmma_f32_16x16x32_f16(false, alo, false, bh,
                                                      (short)0, acc[j], false, false);
    }
  }
  __syncthreads();   // fragment buffer dead; alias it for LN partials below
  float* part  = (float*)smem;            // [wv][row][2] : 16*16*2 floats
  float* stats = (float*)(smem + 2048);   // [row][2]     : mean, inv-std

  // ---- epilogue: o = attn[b, w, l0+m] * (acc + b_v[n]) + enc[row, n] -------
  float att[8];
  #pragma unroll
  for (int i = 0; i < 8; ++i)
    att[i] = attn[((b * NH_ + w) * L_) + l0 + i + 8 * mrow];

  float o[4][8];
  float s_part[8], q_part[8];
  #pragma unroll
  for (int i = 0; i < 8; ++i) { s_part[i] = 0.f; q_part[i] = 0.f; }
  #pragma unroll
  for (int j = 0; j < 4; ++j) {
    int n = w * 64 + j * 16 + (lane & 15);
    float bv = b_v[n];
    #pragma unroll
    for (int i = 0; i < 8; ++i) {
      int mm = i + 8 * mrow;
      float val = att[i] * (acc[j][i] + bv) + enc[(r0 + mm) * DM_ + n];
      o[j][i] = val;
      s_part[i] += val;
      q_part[i] += val * val;
    }
  }
  // reduce across the 16 lanes holding the same row set
  #pragma unroll
  for (int off = 1; off < 16; off <<= 1) {
    #pragma unroll
    for (int i = 0; i < 8; ++i) {
      s_part[i] += __shfl_xor(s_part[i], off, 32);
      q_part[i] += __shfl_xor(q_part[i], off, 32);
    }
  }
  if ((lane & 15) == 0) {
    #pragma unroll
    for (int i = 0; i < 8; ++i) {
      int mm = i + 8 * mrow;
      part[(w * 16 + mm) * 2 + 0] = s_part[i];
      part[(w * 16 + mm) * 2 + 1] = q_part[i];
    }
  }
  __syncthreads();
  if (tid < 16) {
    float ss = 0.f, qq = 0.f;
    #pragma unroll
    for (int wv = 0; wv < 16; ++wv) {
      ss += part[(wv * 16 + tid) * 2 + 0];
      qq += part[(wv * 16 + tid) * 2 + 1];
    }
    float mean = ss * (1.0f / 1024.0f);
    float var  = qq * (1.0f / 1024.0f) - mean * mean;
    stats[tid * 2 + 0] = mean;
    stats[tid * 2 + 1] = rsqrtf(var + 1e-5f);
  }
  __syncthreads();
  #pragma unroll
  for (int j = 0; j < 4; ++j) {
    int n = w * 64 + j * 16 + (lane & 15);
    float g = ln_g[n], bb = ln_b[n];
    #pragma unroll
    for (int i = 0; i < 8; ++i) {
      int mm = i + 8 * mrow;
      float mean = stats[mm * 2 + 0], inv = stats[mm * 2 + 1];
      out[(r0 + mm) * DM_ + n] = (o[j][i] - mean) * inv * g + bb;
    }
  }
}

// ---------------------------------------------------------------------------
extern "C" void kernel_launch(void* const* d_in, const int* in_sizes, int n_in,
                              void* d_out, int out_size, void* d_ws,
                              size_t ws_size, hipStream_t stream) {
  const float* enc  = (const float*)d_in[0];
  const float* w_q  = (const float*)d_in[1];
  const float* b_q  = (const float*)d_in[2];
  const float* w_k  = (const float*)d_in[3];
  const float* b_k  = (const float*)d_in[4];
  const float* w_v  = (const float*)d_in[5];
  const float* b_v  = (const float*)d_in[6];
  const float* ln_g = (const float*)d_in[7];
  const float* ln_b = (const float*)d_in[8];
  float* out = (float*)d_out;

  char* ws = (char*)d_ws;
  float*    q      = (float*)(ws);                               // 64 KB
  float*    wq_eff = (float*)(ws + 65536);                       // 1 MB
  float*    cterm  = (float*)(ws + 65536 + 1048576);             // 1 KB
  float*    scores = (float*)(ws + 65536 + 1048576 + 1024);      // 4 MB
  _Float16* bhi    = (_Float16*)(ws + 65536 + 1048576 + 1024 + 4194304); // 2 MB
  _Float16* blo    = bhi + 1048576;                              // 2 MB

  k0_convert_wv<<<4096, 256, 0, stream>>>(w_v, bhi, blo);
  k1_q<<<dim3(32, 16), 256, 0, stream>>>(enc, w_q, b_q, q);
  k2_wqeff<<<dim3(4, 16, 16), 256, 0, stream>>>(w_k, b_k, q, wq_eff, cterm);
  k3_scores<<<dim3(512, 16), 256, 0, stream>>>(enc, wq_eff, cterm, scores);
  k4_logsoftmax<<<256, 256, 0, stream>>>(scores);
  k5_vgemm_combine_ln<<<4096, 512, 0, stream>>>(enc, bhi, blo, b_v, scores,
                                                ln_g, ln_b, out);
}
